// Edge_denoise_88476326297715
// MI455X (gfx1250) — matrix operand order repaired and checked by reading, not compiled, not run on hardware
//
#include <hip/hip_runtime.h>

// ---------------------------------------------------------------------------
// MI455X / gfx1250 implementation of the EGNN "Edge_denoise" reference.
// All heavy GEMMs (per-edge MLPs of every GCL layer) run through
// v_wmma_f32_16x16x32_bf16 (wave32 WMMA, f32 accumulate).
// Round 2: LDS-staged A operand (unified [h_r|h_c|rad|attr] row per edge),
// paired 32-bit ds gathers, and 4-way grouped B-fragment loads so one
// s_wait_loadcnt covers 4 WMMAs.
// ---------------------------------------------------------------------------

typedef __attribute__((ext_vector_type(16))) __bf16 v16bf;
typedef __attribute__((ext_vector_type(8)))  float  v8f;

#define NNODE  128
#define HDIM   256
#define NTOT   1024
#define EFULL  131072
#define EFOC   8192
#define ESP    2048
#define NFOCAL 48
#define BSZ    8
#define OUTDIM 64
#define KP     800          // padded K for the edge0 GEMM staging row

__device__ __forceinline__ unsigned short f2bf(float f) {
  unsigned int u = __float_as_uint(f);
  unsigned int r = u + 0x7FFFu + ((u >> 16) & 1u);   // round-to-nearest-even
  return (unsigned short)(r >> 16);
}
__device__ __forceinline__ float bf2f(unsigned short h) {
  return __uint_as_float(((unsigned int)h) << 16);
}
__device__ __forceinline__ float siluf(float v) { return v * (1.0f / (1.0f + __expf(-v))); }
__device__ __forceinline__ float sigmf(float v) { return 1.0f / (1.0f + __expf(-v)); }

union AFrag { v16bf v; unsigned int ui[8]; unsigned short us[16]; };
union BFrag { v16bf v; unsigned long long d[4]; unsigned short us[16]; };

// ---------------------------------------------------------------------------
// Pack fp32 weight matrix W (K x N, row-major) into bf16 WMMA B-fragments.
// B order used here (contiguous k, as in the A gather): lane holds column n = lane&15; K-half selected
// by lane>>4; element e maps to k_local = e + 16*(lane>>4). Tiles are stored
// as [tile][lane][e] so a lane fetches its 32 bytes contiguously.
// ---------------------------------------------------------------------------
__global__ void pack_b_kernel(const float* __restrict__ W, int K, int N,
                              unsigned short* __restrict__ dst) {
  int tile = blockIdx.x;                // kt*(N/16) + nt
  int kt = tile / (N / 16);
  int nt = tile % (N / 16);
  int lane = threadIdx.x >> 4;          // 0..31
  int e    = threadIdx.x & 15;          // 0..15
  int k = kt * 32 + (e + 16 * (lane >> 4));
  int n = nt * 16 + (lane & 15);
  float v = (k < K) ? W[(size_t)k * N + n] : 0.0f;
  dst[((size_t)tile * 32 + lane) * 16 + e] = f2bf(v);
}

// ---------------------------------------------------------------------------
// Generic LDS->WMMA GEMM: A is 16 x (nkt*32) bf16 rows in LDS with `stride`
// elements per row; produces 16x256 into acc[16].
// A fragment gather uses paired (32-bit) ds loads: elements (2v,2v+1) are
// consecutive k. B fragments are fetched 4 tiles at a time so the scheduler
// can clause 8 b128 loads per wait.
// ---------------------------------------------------------------------------
__device__ __forceinline__ void gemm_tiles(const unsigned short* __restrict__ s_a,
                                           int stride, int nkt,
                                           const unsigned short* __restrict__ wp,
                                           v8f* acc, int lane) {
  int g  = lane >> 4;
  int ml = lane & 15;
  const unsigned long long* wb = (const unsigned long long*)wp;
  for (int kt = 0; kt < nkt; kt++) {
    AFrag A;
#pragma unroll
    for (int v = 0; v < 8; v++) {
      int kp = kt * 32 + 2 * v + 16 * g;
      A.ui[v] = *(const unsigned int*)(s_a + ml * stride + kp);
    }
#pragma unroll
    for (int nt4 = 0; nt4 < 4; nt4++) {
      const unsigned long long* bp = wb + (size_t)(kt * 16 + nt4 * 4) * 128 + lane * 4;
      BFrag Bf[4];
#pragma unroll
      for (int j = 0; j < 4; j++) {
        Bf[j].d[0] = bp[j * 128 + 0];
        Bf[j].d[1] = bp[j * 128 + 1];
        Bf[j].d[2] = bp[j * 128 + 2];
        Bf[j].d[3] = bp[j * 128 + 3];
      }
#pragma unroll
      for (int j = 0; j < 4; j++) {
        acc[nt4 * 4 + j] = __builtin_amdgcn_wmma_f32_16x16x32_bf16(
            false, A.v, false, Bf[j].v, (short)0, acc[nt4 * 4 + j], false, false);
      }
    }
  }
}

__device__ __forceinline__ void zero_acc(v8f* acc) {
#pragma unroll
  for (int t = 0; t < 16; t++)
#pragma unroll
    for (int r = 0; r < 8; r++) acc[t][r] = 0.0f;
}

__device__ __forceinline__ void store_silu_bf(const v8f* acc, const float* __restrict__ bias,
                                              unsigned short* __restrict__ s_dst, int lane) {
  int g = lane >> 4, nl = lane & 15;
#pragma unroll
  for (int nt = 0; nt < 16; nt++) {
    float bv = bias[nt * 16 + nl];
#pragma unroll
    for (int r = 0; r < 8; r++) {
      float v = siluf(acc[nt][r] + bv);
      s_dst[(r + 8 * g) * HDIM + nt * 16 + nl] = f2bf(v);
    }
  }
}

__device__ __forceinline__ void store_silu_f32(const v8f* acc, const float* __restrict__ bias,
                                               float* __restrict__ s_dst, int lane) {
  int g = lane >> 4, nl = lane & 15;
#pragma unroll
  for (int nt = 0; nt < 16; nt++) {
    float bv = bias[nt * 16 + nl];
#pragma unroll
    for (int r = 0; r < 8; r++) {
      float v = siluf(acc[nt][r] + bv);
      s_dst[(r + 8 * g) * HDIM + nt * 16 + nl] = v;
    }
  }
}

// ---------------------------------------------------------------------------
// Fused GCL edge kernel: 1 wave handles 16 edges.
//   m1 = silu(W0 [h_r|h_c|radial|attr] + b0)       (WMMA, K=nkt0*32)
//   m  = silu(W1 m1 + b1)                          (WMMA, K=256)
//   m *= sigmoid(att(m)) (opt) * em (opt)          -> ef_out (bf16)
//   c1 = silu(Wc0 m + bc0)                         (WMMA, K=256)
//   trans = diff_n * tanh(wc1.c1 + bc1) * 30       -> trans_out
// mode: 0 = full graph (indices analytic, attr = ef),
//       1 = edge list with attr (focal),
//       2 = edge list, attr == radial (sparse, K=514)
// ---------------------------------------------------------------------------
__global__ __launch_bounds__(32)
void gcl_edge_kernel(const float* __restrict__ x,
                     const unsigned short* __restrict__ hbf,
                     const unsigned short* __restrict__ attr,
                     const int* __restrict__ rowIdx, const int* __restrict__ colIdx,
                     const float* __restrict__ em,
                     const unsigned short* __restrict__ w0p, int nkt0,
                     const float* __restrict__ b0,
                     const unsigned short* __restrict__ w1p, const float* __restrict__ b1,
                     const float* __restrict__ attw, const float* __restrict__ attb, int hasAtt,
                     const unsigned short* __restrict__ wc0p, const float* __restrict__ bc0,
                     const float* __restrict__ wc1, const float* __restrict__ wc1b,
                     unsigned short* __restrict__ ef_out,
                     float* __restrict__ trans_out,
                     int E, int mode) {
  __shared__ float s_rad[16], s_dnx[16], s_dny[16], s_dnz[16], s_em[16], s_scale[16];
  __shared__ int s_row[16], s_col[16];
  __shared__ unsigned short s_ain[16 * KP];    // unified A rows [h_r|h_c|rad|attr|pad]
  __shared__ unsigned short s_mb[16 * HDIM];   // bf16 staging (A operand, GEMM2/3)
  __shared__ float s_mf[16 * HDIM];            // fp32 staging (dots)

  int lane = threadIdx.x;
  int e0 = blockIdx.x * 16;

  // ---- per-edge geometry + radial / zero padding of A rows ----------------
  if (lane < 16) {
    int e = e0 + lane;
    int row, col;
    if (mode == 0) {
      int b = e >> 14;            // / (128*128)
      int rem = e & 16383;
      row = b * NNODE + (rem >> 7);
      col = b * NNODE + (rem & 127);
    } else {
      row = rowIdx[e];
      col = colIdx[e];
    }
    float dx = x[row * 3 + 0] - x[col * 3 + 0];
    float dy = x[row * 3 + 1] - x[col * 3 + 1];
    float dz = x[row * 3 + 2] - x[col * 3 + 2];
    float rad = dx * dx + dy * dy + dz * dz;
    float inv = 1.0f / (sqrtf(rad) + 1.0f);
    s_rad[lane] = rad;
    s_dnx[lane] = dx * inv; s_dny[lane] = dy * inv; s_dnz[lane] = dz * inv;
    s_row[lane] = row; s_col[lane] = col;
    s_em[lane] = em ? em[e] : 1.0f;
    unsigned short rb = f2bf(rad);
    s_ain[lane * KP + 512] = rb;
    if (mode == 2) {
      s_ain[lane * KP + 513] = rb;
      for (int i = 514; i < 544; i++) s_ain[lane * KP + i] = 0;
    } else {
      for (int i = 769; i < KP; i++) s_ain[lane * KP + i] = 0;
    }
  }
  __syncthreads();

  // ---- stage h rows (coalesced b128): lanes 0-15 -> h[row], 16-31 -> h[col]
  {
    int r = lane & 15;
    int node = (lane < 16) ? s_row[r] : s_col[r];
    const uint4* src = (const uint4*)(hbf + (size_t)node * HDIM);
    uint4* dst = (uint4*)(s_ain + r * KP + ((lane < 16) ? 0 : 256));
#pragma unroll
    for (int i = 0; i < 32; i++) dst[i] = src[i];
  }
  // ---- stage attr rows (contiguous 16x256 block) into columns 513..768 ----
  if (mode != 2) {
    int r = lane & 15, half = lane >> 4;
    const unsigned short* src = attr + (size_t)(e0 + r) * HDIM + half * 128;
    unsigned short* dst = s_ain + r * KP + 513 + half * 128;
    for (int i = 0; i < 64; i++) {
      unsigned int v = *(const unsigned int*)(src + 2 * i);
      dst[2 * i]     = (unsigned short)(v & 0xFFFFu);
      dst[2 * i + 1] = (unsigned short)(v >> 16);
    }
  }
  __syncthreads();

  v8f acc[16];

  // -------- GEMM1: edge0 over K = [h_row|h_col|radial|attr] ----------------
  zero_acc(acc);
  gemm_tiles(s_ain, KP, nkt0, w0p, acc, lane);
  __syncthreads();
  store_silu_bf(acc, b0, s_mb, lane);     // m1 (bf16) to LDS
  __syncthreads();

  // -------- GEMM2: edge1 ---------------------------------------------------
  zero_acc(acc);
  gemm_tiles(s_mb, HDIM, 8, w1p, acc, lane);
  __syncthreads();
  store_silu_f32(acc, b1, s_mf, lane);    // m (fp32) to LDS
  __syncthreads();

  // -------- attention + mask scale -----------------------------------------
  if (lane < 16) {
    float sc = s_em[lane];
    if (hasAtt) {
      float a = attb[0];
      for (int k = 0; k < HDIM; k++) a += s_mf[lane * HDIM + k] * attw[k];
      sc *= sigmf(a);
    }
    s_scale[lane] = sc;
  }
  __syncthreads();
  for (int idx = lane; idx < 16 * HDIM; idx += 32) {
    float v = s_mf[idx] * s_scale[idx >> 8];
    s_mb[idx] = f2bf(v);
  }
  __syncthreads();
  // final m -> global ef_out (bf16)
  {
    const uint4* src = (const uint4*)s_mb;
    uint4* dst = (uint4*)(ef_out + (size_t)e0 * HDIM);
    for (int i = lane; i < 16 * HDIM / 8; i += 32) dst[i] = src[i];
  }

  // -------- GEMM3: coord0 --------------------------------------------------
  zero_acc(acc);
  gemm_tiles(s_mb, HDIM, 8, wc0p, acc, lane);
  __syncthreads();
  store_silu_f32(acc, bc0, s_mf, lane);   // c1 (fp32) to LDS
  __syncthreads();

  // -------- coord1 + trans -------------------------------------------------
  if (lane < 16) {
    float a = wc1b[0];
    for (int k = 0; k < HDIM; k++) a += s_mf[lane * HDIM + k] * wc1[k];
    float s = tanhf(a) * 30.0f;
    int e = e0 + lane;
    trans_out[e * 3 + 0] = s_dnx[lane] * s;
    trans_out[e * 3 + 1] = s_dny[lane] * s;
    trans_out[e * 3 + 2] = s_dnz[lane] * s;
  }
}

// ---------------------------------------------------------------------------
// Segment-sum kernels
// ---------------------------------------------------------------------------
// Full graph: edges for node v are contiguous (e = v*128 + j) -> plain reduce.
__global__ __launch_bounds__(256)
void full_reduce_kernel(const unsigned short* __restrict__ m, const float* __restrict__ trans,
                        float* __restrict__ agg, const float* __restrict__ x_in,
                        float* __restrict__ x_out) {
  int v = blockIdx.x, f = threadIdx.x;
  size_t base = (size_t)v * NNODE;
  float s = 0.0f;
  for (int j = 0; j < NNODE; j++) s += bf2f(m[(base + j) * HDIM + f]);
  agg[v * HDIM + f] = s;
  if (f < 3) {
    float t = 0.0f;
    for (int j = 0; j < NNODE; j++) t += trans[(base + j) * 3 + f];
    x_out[v * 3 + f] = x_in[v * 3 + f] + t;
  }
}

__global__ __launch_bounds__(256)
void init_agg_x_kernel(float* __restrict__ agg, const float* __restrict__ x_in,
                       float* __restrict__ x_out) {
  int v = blockIdx.x, f = threadIdx.x;
  agg[v * HDIM + f] = 0.0f;
  if (f < 3) x_out[v * 3 + f] = x_in[v * 3 + f];
}

__global__ __launch_bounds__(256)
void scatter_kernel(const unsigned short* __restrict__ m, const float* __restrict__ trans,
                    const int* __restrict__ rowIdx, float* __restrict__ agg,
                    float* __restrict__ x_out, int E) {
  int t = blockIdx.x * blockDim.x + threadIdx.x;
  if (t >= E * HDIM) return;
  int e = t >> 8, f = t & 255;
  int r = rowIdx[e];
  atomicAdd(&agg[r * HDIM + f], bf2f(m[t]));
  if (f < 3) atomicAdd(&x_out[r * 3 + f], trans[e * 3 + f]);
}

// ---------------------------------------------------------------------------
// Node-side MLPs (small: ~1% of FLOPs, scalar VALU)
// ---------------------------------------------------------------------------
__global__ __launch_bounds__(256)
void node_mlp1_kernel(const float* __restrict__ h, const float* __restrict__ agg,
                      const float* __restrict__ w, const float* __restrict__ b,
                      float* __restrict__ t) {
  int v = blockIdx.x, f = threadIdx.x;
  float a = b[f];
  for (int k = 0; k < HDIM; k++) a += h[v * HDIM + k] * w[k * HDIM + f];
  for (int k = 0; k < HDIM; k++) a += agg[v * HDIM + k] * w[(HDIM + k) * HDIM + f];
  t[v * HDIM + f] = siluf(a);
}

__global__ __launch_bounds__(256)
void node_mlp2_kernel(float* __restrict__ h, const float* __restrict__ t,
                      const float* __restrict__ w, const float* __restrict__ b,
                      const float* __restrict__ nm, unsigned short* __restrict__ hbf,
                      float* __restrict__ x) {
  int v = blockIdx.x, f = threadIdx.x;
  float a = b[f];
  for (int k = 0; k < HDIM; k++) a += t[v * HDIM + k] * w[k * HDIM + f];
  float hv = (h[v * HDIM + f] + a) * nm[v];
  h[v * HDIM + f] = hv;
  hbf[v * HDIM + f] = f2bf(hv);
  if (f < 3) x[v * 3 + f] *= nm[v];
}

// ---------------------------------------------------------------------------
// Embedding / init kernels
// ---------------------------------------------------------------------------
__global__ __launch_bounds__(256)
void embed1_kernel(const float* __restrict__ feat, const float* __restrict__ wf,
                   const float* __restrict__ bfp, const int* __restrict__ vocab,
                   const float* __restrict__ vemb, float* __restrict__ hcat) {
  int v = blockIdx.x, f = threadIdx.x;
  float a = bfp[f];
  for (int k = 0; k < 16; k++) a += feat[v * 16 + k] * wf[k * HDIM + f];
  hcat[v * 512 + f] = a;
  hcat[v * 512 + 256 + f] = vemb[(size_t)vocab[v] * HDIM + f];
}

__global__ __launch_bounds__(256)
void embed2_kernel(const float* __restrict__ hcat, const float* __restrict__ w,
                   const float* __restrict__ b, float* __restrict__ h,
                   unsigned short* __restrict__ hbf) {
  int v = blockIdx.x, f = threadIdx.x;
  float a = b[f];
  for (int k = 0; k < 512; k++) a += hcat[v * 512 + k] * w[k * HDIM + f];
  h[v * HDIM + f] = a;
  hbf[v * HDIM + f] = f2bf(a);
}

__global__ void val_kernel(const float* __restrict__ adj, float* __restrict__ val) {
  int v = blockIdx.x * blockDim.x + threadIdx.x;
  if (v < NTOT) {
    float s = 0.0f;
    for (int j = 0; j < NNODE; j++) s += adj[(size_t)v * NNODE + j];
    val[v] = s;
  }
}

__global__ __launch_bounds__(256)
void ef_init_kernel(const float* __restrict__ x, const float* __restrict__ adj,
                    const float* __restrict__ we, const float* __restrict__ be,
                    unsigned short* __restrict__ ef) {
  size_t t = (size_t)blockIdx.x * blockDim.x + threadIdx.x;
  int e = (int)(t >> 8), f = (int)(t & 255);
  int b = e >> 14, rem = e & 16383;
  int row = b * NNODE + (rem >> 7), col = b * NNODE + (rem & 127);
  float dx = x[row * 3 + 0] - x[col * 3 + 0];
  float dy = x[row * 3 + 1] - x[col * 3 + 1];
  float dz = x[row * 3 + 2] - x[col * 3 + 2];
  float dist = dx * dx + dy * dy + dz * dz;
  float a = be[f] + dist * we[f] + adj[e] * we[HDIM + f];
  ef[t] = f2bf(a);
}

__global__ __launch_bounds__(256)
void gather_efoc_kernel(const unsigned short* __restrict__ ef_full,
                        const int* __restrict__ esf, unsigned short* __restrict__ out) {
  size_t t = (size_t)blockIdx.x * blockDim.x + threadIdx.x;
  int e = (int)(t >> 8), f = (int)(t & 255);
  int a0 = esf[e], a1 = esf[EFOC + e];
  size_t fe = (size_t)a0 * NNODE + (a1 & 127);  // ef_full[e0//n, e0%n, e1%n] flat
  out[t] = ef_full[fe * HDIM + f];
}

// ---------------------------------------------------------------------------
// Heads
// ---------------------------------------------------------------------------
__global__ __launch_bounds__(256)
void focal_head1_kernel(const float* __restrict__ h, const float* __restrict__ val,
                        const int* __restrict__ cand, const float* __restrict__ w,
                        const float* __restrict__ b, float* __restrict__ t) {
  int i = blockIdx.x, f = threadIdx.x;
  int node = cand[i];
  float a = b[f];
  for (int k = 0; k < HDIM; k++) a += h[node * HDIM + k] * w[k * HDIM + f];
  a += val[node] * w[HDIM * HDIM + f];   // row 256 of (257,256)
  t[i * HDIM + f] = siluf(a);
}

__global__ void focal_head2_kernel(const float* __restrict__ t, const float* __restrict__ w,
                                   const float* __restrict__ b, float* __restrict__ out) {
  int i = blockIdx.x * blockDim.x + threadIdx.x;
  if (i < NFOCAL) {
    float a = b[0];
    for (int k = 0; k < HDIM; k++) a += t[i * HDIM + k] * w[k];
    out[i] = sigmf(a);
  }
}

__global__ __launch_bounds__(256)
void edge_head1_kernel(const float* __restrict__ h, const float* __restrict__ x,
                       const unsigned short* __restrict__ ef_full,
                       const int* __restrict__ real_focal, const float* __restrict__ w,
                       const float* __restrict__ b, float* __restrict__ t) {
  int r = blockIdx.x;          // 0..1023 = (sample, attach j)
  int f = threadIdx.x;
  int bs_ = r >> 7, j = r & 127;
  int rf = real_focal[bs_];
  int fb = rf / NNODE;
  int attach = fb * NNODE + j;
  float a = b[f];
  for (int k = 0; k < HDIM; k++) a += h[rf * HDIM + k] * w[k * HDIM + f];
  for (int k = 0; k < HDIM; k++)
    a += bf2f(ef_full[((size_t)rf * NNODE + j) * HDIM + k]) * w[(HDIM + k) * HDIM + f];
  for (int k = 0; k < HDIM; k++) a += h[attach * HDIM + k] * w[(2 * HDIM + k) * HDIM + f];
  float dx = x[attach * 3 + 0] - x[rf * 3 + 0];
  float dy = x[attach * 3 + 1] - x[rf * 3 + 1];
  float dz = x[attach * 3 + 2] - x[rf * 3 + 2];
  a += (dx * dx + dy * dy + dz * dz) * w[3 * HDIM * HDIM + f];
  t[r * HDIM + f] = siluf(a);
}

__global__ void edge_head2_kernel(const float* __restrict__ t, const float* __restrict__ w,
                                  const float* __restrict__ b, float* __restrict__ out) {
  int i = blockIdx.x * blockDim.x + threadIdx.x;
  if (i < NTOT) {
    float a = b[0];
    for (int k = 0; k < HDIM; k++) a += t[i * HDIM + k] * w[k];
    out[NFOCAL + i] = a;   // no sigmoid
  }
}

__global__ __launch_bounds__(256)
void node_head1_kernel(const float* __restrict__ h, const int* __restrict__ pidx,
                       const float* __restrict__ w, const float* __restrict__ b,
                       float* __restrict__ t) {
  int i = blockIdx.x, f = threadIdx.x;
  int node = i * NNODE + pidx[i];
  float a = b[f];
  for (int k = 0; k < HDIM; k++) a += h[node * HDIM + k] * w[k * HDIM + f];
  t[i * HDIM + f] = siluf(a);
}

__global__ void node_head2_kernel(const float* __restrict__ t, const float* __restrict__ w,
                                  const float* __restrict__ b, float* __restrict__ out) {
  int i = blockIdx.x, f = threadIdx.x;  // 8 x 64
  float a = b[f];
  for (int k = 0; k < HDIM; k++) a += t[i * HDIM + k] * w[k * OUTDIM + f];
  out[NFOCAL + NTOT + i * OUTDIM + f] = a;
}

// ---------------------------------------------------------------------------
// Host orchestration
// ---------------------------------------------------------------------------
// Assumed input flattening (setup_inputs insertion order, params pytree in
// insertion order, each leaf one entry):
enum {
  IN_NODE_FEAT = 0, IN_NODE_VOCAB, IN_NODE_POS, IN_NODE_MASK, IN_EDGE_MASK,
  IN_SEARCH_ADJ, IN_EDGE_SEARCH, IN_SPARSE_EDGES, IN_FOCAL_CAND, IN_REAL_FOCAL,
  IN_PREDICT_IDX,
  P_FEAT_W, P_FEAT_B, P_VOCAB, P_EDGEEMB_W, P_EDGEEMB_B, P_NODEEMB_W, P_NODEEMB_B,
  P_GCL0   // 14 leaves per GCL x 8 GCLs (3 full, 3 focal, edge, denoise)
};
#define P_FH (P_GCL0 + 14 * 8)   // focal_head l0.w,l0.b,l1.w,l1.b
#define P_EH (P_FH + 4)
#define P_NH (P_EH + 4)

struct GclP {
  const float *e0w, *e0b, *e1w, *e1b, *attw, *attb, *c0w, *c0b, *c1w, *c1b,
              *n0w, *n0b, *n1w, *n1b;
};

extern "C" void kernel_launch(void* const* d_in, const int* in_sizes, int n_in,
                              void* d_out, int out_size, void* d_ws, size_t ws_size,
                              hipStream_t stream) {
  (void)in_sizes; (void)n_in; (void)out_size; (void)ws_size;
  auto F = [&](int i) { return (const float*)d_in[i]; };
  auto I = [&](int i) { return (const int*)d_in[i]; };

  GclP gcl[8];
  for (int gi = 0; gi < 8; gi++) {
    int b = P_GCL0 + 14 * gi;
    gcl[gi] = { F(b+0), F(b+1), F(b+2), F(b+3), F(b+4), F(b+5), F(b+6),
                F(b+7), F(b+8), F(b+9), F(b+10), F(b+11), F(b+12), F(b+13) };
  }

  // ---- workspace carve-up (≈155 MB; sized to stay resident in 192MB L2) ----
  char* ws = (char*)d_ws;
  size_t off = 0;
  auto alloc = [&](size_t bytes) { size_t o = off; off = (off + bytes + 255) & ~(size_t)255; return o; };
  float*          h     = (float*)(ws + alloc((size_t)NTOT * HDIM * 4));
  unsigned short* hbf   = (unsigned short*)(ws + alloc((size_t)NTOT * HDIM * 2));
  float*          xA    = (float*)(ws + alloc((size_t)NTOT * 3 * 4));
  float*          xB    = (float*)(ws + alloc((size_t)NTOT * 3 * 4));
  float*          agg   = (float*)(ws + alloc((size_t)NTOT * HDIM * 4));
  float*          tbuf  = (float*)(ws + alloc((size_t)NTOT * HDIM * 4));
  float*          hcat  = (float*)(ws + alloc((size_t)NTOT * 512 * 4));
  float*          val   = (float*)(ws + alloc((size_t)NTOT * 4));
  float*          trans = (float*)(ws + alloc((size_t)EFULL * 3 * 4));
  unsigned short* buf0  = (unsigned short*)(ws + alloc((size_t)EFULL * HDIM * 2));
  unsigned short* buf1  = (unsigned short*)(ws + alloc((size_t)EFULL * HDIM * 2));
  unsigned short* foc0  = (unsigned short*)(ws + alloc((size_t)EFOC * HDIM * 2));
  unsigned short* foc1  = (unsigned short*)(ws + alloc((size_t)EFOC * HDIM * 2));
  unsigned short* pkE0[8]; unsigned short* pkE1[8]; unsigned short* pkC0[8];
  for (int gi = 0; gi < 8; gi++) {
    int nkt0 = (gi < 6) ? 25 : 17;
    pkE0[gi] = (unsigned short*)(ws + alloc((size_t)nkt0 * 16 * 32 * 16 * 2));
    pkE1[gi] = (unsigned short*)(ws + alloc((size_t)8 * 16 * 32 * 16 * 2));
    pkC0[gi] = (unsigned short*)(ws + alloc((size_t)8 * 16 * 32 * 16 * 2));
  }

  // ---- pack all WMMA B operands (deterministic, repacked every call) ------
  for (int gi = 0; gi < 8; gi++) {
    int nkt0 = (gi < 6) ? 25 : 17;
    int K0   = (gi < 6) ? 769 : 514;
    pack_b_kernel<<<nkt0 * 16, 512, 0, stream>>>(gcl[gi].e0w, K0, HDIM, pkE0[gi]);
    pack_b_kernel<<<8 * 16, 512, 0, stream>>>(gcl[gi].e1w, HDIM, HDIM, pkE1[gi]);
    pack_b_kernel<<<8 * 16, 512, 0, stream>>>(gcl[gi].c0w, HDIM, HDIM, pkC0[gi]);
  }

  // ---- embeddings / val / ef init ----------------------------------------
  embed1_kernel<<<NTOT, 256, 0, stream>>>(F(IN_NODE_FEAT), F(P_FEAT_W), F(P_FEAT_B),
                                          I(IN_NODE_VOCAB), F(P_VOCAB), hcat);
  embed2_kernel<<<NTOT, 256, 0, stream>>>(hcat, F(P_NODEEMB_W), F(P_NODEEMB_B), h, hbf);
  val_kernel<<<(NTOT + 255) / 256, 256, 0, stream>>>(F(IN_SEARCH_ADJ), val);
  ef_init_kernel<<<EFULL, 256, 0, stream>>>(F(IN_NODE_POS), F(IN_SEARCH_ADJ),
                                            F(P_EDGEEMB_W), F(P_EDGEEMB_B), buf0);

  const float* nm = F(IN_NODE_MASK);
  const float* xcur = F(IN_NODE_POS);
  float* xnxt = xA;

  auto run_gcl = [&](int gi, int mode, int E, const int* rI, const int* cI,
                     const unsigned short* attr, const float* em, unsigned short* mOut) {
    int nkt0 = (gi < 6) ? 25 : 17;
    int hasAtt = (gi < 3) ? 1 : 0;
    gcl_edge_kernel<<<E / 16, 32, 0, stream>>>(
        xcur, hbf, attr, rI, cI, em,
        pkE0[gi], nkt0, gcl[gi].e0b, pkE1[gi], gcl[gi].e1b,
        gcl[gi].attw, gcl[gi].attb, hasAtt,
        pkC0[gi], gcl[gi].c0b, gcl[gi].c1w, gcl[gi].c1b,
        mOut, trans, E, mode);
    if (mode == 0) {
      full_reduce_kernel<<<NTOT, 256, 0, stream>>>(mOut, trans, agg, xcur, xnxt);
    } else {
      init_agg_x_kernel<<<NTOT, 256, 0, stream>>>(agg, xcur, xnxt);
      int tot = E * HDIM;
      scatter_kernel<<<(tot + 255) / 256, 256, 0, stream>>>(mOut, trans, rI, agg, xnxt, E);
    }
    node_mlp1_kernel<<<NTOT, 256, 0, stream>>>(h, agg, gcl[gi].n0w, gcl[gi].n0b, tbuf);
    node_mlp2_kernel<<<NTOT, 256, 0, stream>>>(h, tbuf, gcl[gi].n1w, gcl[gi].n1b, nm, hbf, xnxt);
    xcur = xnxt;
    xnxt = (xcur == xA) ? xB : xA;
  };

  // ---- 3 full-graph GCLs (attention + edge mask) --------------------------
  run_gcl(0, 0, EFULL, nullptr, nullptr, buf0, F(IN_EDGE_MASK), buf1);
  run_gcl(1, 0, EFULL, nullptr, nullptr, buf1, F(IN_EDGE_MASK), buf0);
  run_gcl(2, 0, EFULL, nullptr, nullptr, buf0, F(IN_EDGE_MASK), buf1);
  unsigned short* ef_full = buf1;

  // ---- 3 focal GCLs over the search edge list -----------------------------
  const int* esf = I(IN_EDGE_SEARCH);
  gather_efoc_kernel<<<EFOC, 256, 0, stream>>>(ef_full, esf, foc0);
  run_gcl(3, 1, EFOC, esf, esf + EFOC, foc0, nullptr, foc1);
  run_gcl(4, 1, EFOC, esf, esf + EFOC, foc1, nullptr, foc0);
  run_gcl(5, 1, EFOC, esf, esf + EFOC, foc0, nullptr, foc1);

  // ---- focal head ---------------------------------------------------------
  float* out = (float*)d_out;
  focal_head1_kernel<<<NFOCAL, 256, 0, stream>>>(h, val, I(IN_FOCAL_CAND),
                                                 F(P_FH + 0), F(P_FH + 1), tbuf);
  focal_head2_kernel<<<1, 64, 0, stream>>>(tbuf, F(P_FH + 2), F(P_FH + 3), out);

  // ---- edge_gcl over sparse edges (attr = radial) -------------------------
  const int* sp = I(IN_SPARSE_EDGES);
  run_gcl(6, 2, ESP, sp, sp + ESP, nullptr, nullptr, foc0);

  // ---- edge head ----------------------------------------------------------
  edge_head1_kernel<<<NTOT, 256, 0, stream>>>(h, xcur, ef_full, I(IN_REAL_FOCAL),
                                              F(P_EH + 0), F(P_EH + 1), tbuf);
  edge_head2_kernel<<<4, 256, 0, stream>>>(tbuf, F(P_EH + 2), F(P_EH + 3), out);

  // ---- denoise_gcl + node head -------------------------------------------
  run_gcl(7, 2, ESP, sp, sp + ESP, nullptr, nullptr, foc0);
  node_head1_kernel<<<BSZ, 256, 0, stream>>>(h, I(IN_PREDICT_IDX), F(P_NH + 0), F(P_NH + 1), tbuf);
  node_head2_kernel<<<BSZ, OUTDIM, 0, stream>>>(tbuf, F(P_NH + 2), F(P_NH + 3), out);
}
